// CAFM_62689342652516
// MI455X (gfx1250) — matrix operand order repaired and checked by reading, not compiled, not run on hardware
//
#include <hip/hip_runtime.h>
#include <hip/hip_bf16.h>
#include <math.h>

#define HEADS 8
#define CPH 8
#define BB 16
#define HWSZ 16384   // 128*128
#define IMG 128

typedef __attribute__((ext_vector_type(16))) _Float16 v16h;
typedef __attribute__((ext_vector_type(8)))  _Float16 v8h;
typedef __attribute__((ext_vector_type(8)))  float    v8f;

// ---------------------------------------------------------------------------
// WMMA fragment loaders (CDNA5 16x16x32 f16, wave32 layouts per ISA 7.12.2).
// All loads are 16B-aligned contiguous pairs -> lower to ds_load_b128.
// A: 16x32, row M = lane&15; halves 0-7 = K {kb..kb+7}, 8-15 = K {16+kb..},
//    kb = (lane<16)?0:8.  Requires lda % 8 == 0.
// B (linear): caller passes lane's 32-K base pointer (16B aligned);
//    halves 0-7 = K {kb..kb+7}, 8-15 = K {16+kb..}, kb = (lane<16)?0:8.
// C/D: v8f, M = (lane<16 ? i : 8+i), N = lane&15
// ---------------------------------------------------------------------------
__device__ __forceinline__ v16h frag_a(const _Float16* __restrict__ A, int lda, int lane) {
  const int row = lane & 15;
  const int kb  = (lane & 16) ? 8 : 0;
  const _Float16* p = A + row * lda + kb;
  v8h a0 = *(const v8h*)(p);
  v8h a1 = *(const v8h*)(p + 16);
  return __builtin_shufflevector(a0, a1, 0, 1, 2, 3, 4, 5, 6, 7,
                                 8, 9, 10, 11, 12, 13, 14, 15);
}

__device__ __forceinline__ v16h frag_b_lin(const _Float16* __restrict__ p, int lane) {
  const int kb = (lane & 16) ? 8 : 0;
  v8h b0 = *(const v8h*)(p + kb);
  v8h b1 = *(const v8h*)(p + kb + 16);
  return __builtin_shufflevector(b0, b1, 0, 1, 2, 3, 4, 5, 6, 7,
                                 8, 9, 10, 11, 12, 13, 14, 15);
}

#define WMMA_F16(a, b, c) \
  __builtin_amdgcn_wmma_f32_16x16x32_f16(false, (a), false, (b), (short)0, (c), false, false)

// ---------------------------------------------------------------------------
// 1) qkv = w_qkv @ y : per-batch GEMM M=192, K=64, N=16384 (WMMA f16)
// ---------------------------------------------------------------------------
__global__ __launch_bounds__(256) void k_qkv_gemm(const float* __restrict__ y,
                                                  const float* __restrict__ w_qkv,
                                                  float* __restrict__ qkv) {
  __shared__ __align__(16) _Float16 wq[192 * 64];   // 24 KB  A: [192][64]
  __shared__ __align__(16) _Float16 yt[128 * 72];   // 18 KB  B^T: [pixel][K], stride 72
  const int tid = threadIdx.x;
  const int b   = blockIdx.x >> 7;            // HWSZ/128 = 128 tiles
  const int n0  = (blockIdx.x & 127) * 128;

  for (int i = tid; i < 192 * 64; i += 256) wq[i] = (_Float16)w_qkv[i];
  for (int i = tid; i < 64 * 128; i += 256) {
    const int c = i >> 7, p = i & 127;
    yt[p * 72 + c] = (_Float16)y[((size_t)b * 64 + c) * HWSZ + n0 + p];
  }
  __syncthreads();

  const int wave = tid >> 5, lane = tid & 31;
  const int col  = lane & 15;
  const int rbase = (lane & 16) ? 8 : 0;
  const _Float16* Bp = &yt[(wave * 16 + col) * 72];

  for (int m = 0; m < 12; ++m) {
    v8f acc = {0.f, 0.f, 0.f, 0.f, 0.f, 0.f, 0.f, 0.f};
#pragma unroll
    for (int kc = 0; kc < 2; ++kc) {
      v16h a  = frag_a(&wq[(m * 16) * 64 + kc * 32], 64, lane);
      v16h bf = frag_b_lin(Bp + kc * 32, lane);
      acc = WMMA_F16(a, bf, acc);
    }
#pragma unroll
    for (int i = 0; i < 8; ++i) {
      const int row = m * 16 + rbase + i;
      qkv[((size_t)b * 192 + row) * HWSZ + n0 + wave * 16 + col] = acc[i];
    }
  }
}

// ---------------------------------------------------------------------------
// 2) depthwise 3x3, pad=1, groups=192 (memory bound, VALU)
// ---------------------------------------------------------------------------
__global__ __launch_bounds__(256) void k_dw(const float* __restrict__ qkv,
                                            const float* __restrict__ w_dw,
                                            float* __restrict__ qkvdw) {
  const size_t idx = (size_t)blockIdx.x * 256 + threadIdx.x;  // B*192*HW
  const int pix = (int)(idx & (HWSZ - 1));
  const size_t bc = idx >> 14;
  const int ch = (int)(bc % 192);
  const int px = pix & (IMG - 1), py = pix >> 7;
  const float* in = qkv + (bc << 14);
  float s = 0.f;
#pragma unroll
  for (int ky = 0; ky < 3; ++ky) {
    const int yy = py + ky - 1;
    if (yy < 0 || yy >= IMG) continue;
#pragma unroll
    for (int kx = 0; kx < 3; ++kx) {
      const int xx = px + kx - 1;
      if (xx < 0 || xx >= IMG) continue;
      s += w_dw[ch * 9 + ky * 3 + kx] * in[yy * IMG + xx];
    }
  }
  qkvdw[idx] = s;
}

// ---------------------------------------------------------------------------
// 3) channel attention stats: per (b,head) L2 norms + 8x8 gram + softmax
// ---------------------------------------------------------------------------
__global__ __launch_bounds__(256) void k_attn(const float* __restrict__ qkvdw,
                                              const float* __restrict__ temperature,
                                              float* __restrict__ attn) {
  const int b = blockIdx.x >> 3;
  const int h = blockIdx.x & 7;
  const float* qb = qkvdw + ((size_t)b * 192 + h * 8) * HWSZ;
  const float* kp = qkvdw + ((size_t)b * 192 + 64 + h * 8) * HWSZ;

  float dot[64], nq2[8], nk2[8];
#pragma unroll
  for (int i = 0; i < 64; ++i) dot[i] = 0.f;
#pragma unroll
  for (int i = 0; i < 8; ++i) { nq2[i] = 0.f; nk2[i] = 0.f; }

  for (int n = threadIdx.x; n < HWSZ; n += 256) {
    float qv[8], kv[8];
#pragma unroll
    for (int c = 0; c < 8; ++c) {
      qv[c] = qb[(size_t)c * HWSZ + n];
      kv[c] = kp[(size_t)c * HWSZ + n];
    }
#pragma unroll
    for (int c = 0; c < 8; ++c) {
      nq2[c] += qv[c] * qv[c];
      nk2[c] += kv[c] * kv[c];
#pragma unroll
      for (int d = 0; d < 8; ++d) dot[c * 8 + d] += qv[c] * kv[d];
    }
  }

  __shared__ float red[80 * 257];
  __shared__ float sums[80];
#pragma unroll
  for (int i = 0; i < 64; ++i) red[i * 257 + threadIdx.x] = dot[i];
#pragma unroll
  for (int i = 0; i < 8; ++i) {
    red[(64 + i) * 257 + threadIdx.x] = nq2[i];
    red[(72 + i) * 257 + threadIdx.x] = nk2[i];
  }
  __syncthreads();
  if (threadIdx.x < 80) {
    float s = 0.f;
    for (int t = 0; t < 256; ++t) s += red[threadIdx.x * 257 + t];
    sums[threadIdx.x] = s;
  }
  __syncthreads();
  if (threadIdx.x < 8) {
    const int c = threadIdx.x;
    const float temp = temperature[h];
    const float nqc = fmaxf(sqrtf(sums[64 + c]), 1e-12f);
    float row[8];
    float mx = -1e30f;
#pragma unroll
    for (int d = 0; d < 8; ++d) {
      const float nkd = fmaxf(sqrtf(sums[72 + d]), 1e-12f);
      row[d] = sums[c * 8 + d] / (nqc * nkd) * temp;
      mx = fmaxf(mx, row[d]);
    }
    float den = 0.f;
#pragma unroll
    for (int d = 0; d < 8; ++d) { row[d] = __expf(row[d] - mx); den += row[d]; }
#pragma unroll
    for (int d = 0; d < 8; ++d)
      attn[(((size_t)b * 8 + h) * 8 + c) * 8 + d] = row[d] / den;
  }
}

// ---------------------------------------------------------------------------
// 4) W_eff[b] = w_proj @ blockdiag(attn[b])  (64x64 per batch, tiny)
// ---------------------------------------------------------------------------
__global__ __launch_bounds__(64) void k_weff(const float* __restrict__ w_proj,
                                             const float* __restrict__ attn,
                                             float* __restrict__ weff) {
  const int b = blockIdx.x, o = threadIdx.x;
  const float* at = attn + (size_t)b * 512;
  for (int h = 0; h < 8; ++h)
    for (int d = 0; d < 8; ++d) {
      float s = 0.f;
#pragma unroll
      for (int c = 0; c < 8; ++c)
        s += w_proj[o * 64 + h * 8 + c] * at[((size_t)h * 8 + c) * 8 + d];
      weff[((size_t)b * 64 + o) * 64 + h * 8 + d] = s;
    }
}

// ---------------------------------------------------------------------------
// 5) dynamic-kernel branch: scrambled flat reshape + fc 24->9 per (pixel,e)
//    f_conv[b, e*9+o, n] = b_fc[o] + sum_c w_fc[o,c]*qkvdw_flat[b][n*192+c*8+e]
// ---------------------------------------------------------------------------
__global__ __launch_bounds__(256) void k_fc_dyn(const float* __restrict__ qkvdw,
                                                const float* __restrict__ w_fc,
                                                const float* __restrict__ b_fc,
                                                float* __restrict__ fconv) {
  __shared__ float wfc[9 * 24];
  if (threadIdx.x < 216) wfc[threadIdx.x] = w_fc[threadIdx.x];
  __syncthreads();
  const size_t idx = (size_t)blockIdx.x * 256 + threadIdx.x;  // B*HW
  const int n = (int)(idx & (HWSZ - 1));
  const int b = (int)(idx >> 14);
  const float* xb = qkvdw + ((size_t)b * 192) * HWSZ + (size_t)n * 192;
  float* outb = fconv + (size_t)b * 72 * HWSZ + n;
  for (int e = 0; e < 8; ++e) {
    float x[24];
#pragma unroll
    for (int c = 0; c < 24; ++c) x[c] = xb[c * 8 + e];
#pragma unroll
    for (int o = 0; o < 9; ++o) {
      float s = b_fc[o];
#pragma unroll
      for (int c = 0; c < 24; ++c) s += wfc[o * 24 + c] * x[c];
      outb[(size_t)(e * 9 + o) * HWSZ] = s;
    }
  }
}

// ---------------------------------------------------------------------------
// 6) grouped conv 3x3, 72->64, groups=8, +bias -> writes outsum
// ---------------------------------------------------------------------------
__global__ __launch_bounds__(256) void k_groupconv(const float* __restrict__ fconv,
                                                   const float* __restrict__ w_dep,
                                                   const float* __restrict__ b_dep,
                                                   float* __restrict__ outsum) {
  const size_t idx = (size_t)blockIdx.x * 256 + threadIdx.x;  // B*64*HW
  const int pix = (int)(idx & (HWSZ - 1));
  const size_t bo = idx >> 14;
  const int o = (int)(bo & 63);
  const int b = (int)(bo >> 6);
  const int g = o >> 3;
  const int px = pix & (IMG - 1), py = pix >> 7;
  const float* in = fconv + ((size_t)b * 72 + g * 9) * HWSZ;
  const float* w = w_dep + o * 81;
  float s = b_dep[o];
  for (int i = 0; i < 9; ++i) {
#pragma unroll
    for (int ky = 0; ky < 3; ++ky) {
      const int yy = py + ky - 1;
      if (yy < 0 || yy >= IMG) continue;
#pragma unroll
      for (int kx = 0; kx < 3; ++kx) {
        const int xx = px + kx - 1;
        if (xx < 0 || xx >= IMG) continue;
        s += w[i * 9 + ky * 3 + kx] * in[(size_t)i * HWSZ + yy * IMG + xx];
      }
    }
  }
  outsum[idx] = s;
}

// ---------------------------------------------------------------------------
// 7) outsum += W_eff[b] @ v : per-batch GEMM M=64, K=64, N=16384 (WMMA f16)
// ---------------------------------------------------------------------------
__global__ __launch_bounds__(256) void k_pv_gemm(const float* __restrict__ qkvdw,
                                                 const float* __restrict__ weff,
                                                 float* __restrict__ outsum) {
  __shared__ __align__(16) _Float16 wl[64 * 64];    // 8 KB
  __shared__ __align__(16) _Float16 vt[128 * 72];   // 18 KB  B^T: [pixel][K]
  const int tid = threadIdx.x;
  const int b   = blockIdx.x >> 7;
  const int n0  = (blockIdx.x & 127) * 128;

  for (int i = tid; i < 64 * 64; i += 256) wl[i] = (_Float16)weff[(size_t)b * 4096 + i];
  for (int i = tid; i < 64 * 128; i += 256) {
    const int c = i >> 7, p = i & 127;
    vt[p * 72 + c] = (_Float16)qkvdw[((size_t)b * 192 + 128 + c) * HWSZ + n0 + p];
  }
  __syncthreads();

  const int wave = tid >> 5, lane = tid & 31;
  const int col  = lane & 15;
  const int rbase = (lane & 16) ? 8 : 0;
  const _Float16* Bp = &vt[(wave * 16 + col) * 72];

  for (int m = 0; m < 4; ++m) {
    v8f acc = {0.f, 0.f, 0.f, 0.f, 0.f, 0.f, 0.f, 0.f};
#pragma unroll
    for (int kc = 0; kc < 2; ++kc) {
      v16h a  = frag_a(&wl[(m * 16) * 64 + kc * 32], 64, lane);
      v16h bf = frag_b_lin(Bp + kc * 32, lane);
      acc = WMMA_F16(a, bf, acc);
    }
#pragma unroll
    for (int i = 0; i < 8; ++i) {
      const int row = m * 16 + rbase + i;
      outsum[((size_t)b * 64 + row) * HWSZ + n0 + wave * 16 + col] += acc[i];
    }
  }
}

// ---------------------------------------------------------------------------
// 8) fuse conv 3x3 (64->64) via implicit-im2col WMMA GEMM, + BN + residual
//    + ReLU.  K reordered to k' = (ky*3+kx)*64 + c so every B fragment is
//    16 channels (two runs of 8, 16 apart) of one halo pixel (channel-innermost halo).
// ---------------------------------------------------------------------------
__global__ __launch_bounds__(256) void k_fuse(const float* __restrict__ outsum,
                                              const float* __restrict__ w_fuse,
                                              const float* __restrict__ yres,
                                              const float* __restrict__ bn_g,
                                              const float* __restrict__ bn_b,
                                              const float* __restrict__ bn_m,
                                              const float* __restrict__ bn_v,
                                              float* __restrict__ out) {
  __shared__ __align__(16) _Float16 wf[64 * 576];     // 72 KB : A [o][k'], k'=r*64+c
  __shared__ __align__(16) _Float16 halo[324 * 72];   // 45.6 KB : [y][x][c], stride 72
  const int tid = threadIdx.x;
  const int b = blockIdx.x >> 6;
  const int t = blockIdx.x & 63;
  const int ty = (t >> 3) * 16, tx = (t & 7) * 16;

  // prefetch the residual tile (consumed only in the epilogue)
  __builtin_prefetch(&yres[((size_t)b * 64 + (tid & 63)) * HWSZ + (size_t)ty * IMG + tx], 0, 1);

  // A: permuted weight tile, LDS-consecutive writes
  for (int i = tid; i < 64 * 576; i += 256) {
    const int o  = i / 576;
    const int kq = i - o * 576;          // kq = r*64 + c
    const int r  = kq >> 6, c = kq & 63;
    wf[i] = (_Float16)w_fuse[o * 576 + c * 9 + r];
  }
  // halo: [y][x][c] with pixel stride 72 (16B-aligned rows)
  for (int i = tid; i < 64 * 324; i += 256) {
    const int c = i / 324;
    const int r = i - c * 324;
    const int hy = r / 18, hx = r - hy * 18;
    const int gy = ty + hy - 1, gx = tx + hx - 1;
    float v = 0.f;
    if (gy >= 0 && gy < IMG && gx >= 0 && gx < IMG)
      v = outsum[((size_t)b * 64 + c) * HWSZ + gy * IMG + gx];
    halo[r * 72 + c] = (_Float16)v;
  }
  __syncthreads();

  const int wave = tid >> 5, lane = tid & 31;
  const int col  = lane & 15;
  const int rbase = (lane & 16) ? 8 : 0;
  const int s0 = wave;       // tile row handled first
  const int s1 = wave + 8;   // tile row handled second

  v8f acc[4][2];
  const v8f vzero = {0.f, 0.f, 0.f, 0.f, 0.f, 0.f, 0.f, 0.f};
#pragma unroll
  for (int m = 0; m < 4; ++m) { acc[m][0] = vzero; acc[m][1] = vzero; }

#pragma unroll
  for (int kc = 0; kc < 18; ++kc) {
    const int r  = kc >> 1;              // tap index: ky*3+kx
    const int ky = r / 3;
    const int kx = r - ky * 3;
    const int co = (kc & 1) * 32;        // channel offset within the tap
    v16h b0 = frag_b_lin(&halo[((s0 + ky) * 18 + col + kx) * 72 + co], lane);
    v16h b1 = frag_b_lin(&halo[((s1 + ky) * 18 + col + kx) * 72 + co], lane);
#pragma unroll
    for (int m = 0; m < 4; ++m) {
      v16h a = frag_a(&wf[(m * 16) * 576 + kc * 32], 576, lane);
      acc[m][0] = WMMA_F16(a, b0, acc[m][0]);
      acc[m][1] = WMMA_F16(a, b1, acc[m][1]);
    }
  }

#pragma unroll
  for (int m = 0; m < 4; ++m) {
#pragma unroll
    for (int j = 0; j < 2; ++j) {
      const int srow = (j == 0) ? s0 : s1;
#pragma unroll
      for (int i = 0; i < 8; ++i) {
        const int o = m * 16 + rbase + i;
        const size_t gidx = ((size_t)b * 64 + o) * HWSZ + (size_t)(ty + srow) * IMG + (tx + col);
        const float scale = bn_g[o] * rsqrtf(bn_v[o] + 1e-5f);
        float val = (acc[m][j][i] - bn_m[o]) * scale + bn_b[o];
        val += yres[gidx];
        out[gidx] = fmaxf(val, 0.f);
      }
    }
  }
}

// ---------------------------------------------------------------------------
extern "C" void kernel_launch(void* const* d_in, const int* in_sizes, int n_in,
                              void* d_out, int out_size, void* d_ws, size_t ws_size,
                              hipStream_t stream) {
  const float* y       = (const float*)d_in[0];
  const float* w_qkv   = (const float*)d_in[1];
  const float* w_dw    = (const float*)d_in[2];
  const float* w_proj  = (const float*)d_in[3];
  const float* w_fc    = (const float*)d_in[4];
  const float* b_fc    = (const float*)d_in[5];
  const float* w_dep   = (const float*)d_in[6];
  const float* b_dep   = (const float*)d_in[7];
  const float* temper  = (const float*)d_in[8];
  const float* w_fuse  = (const float*)d_in[9];
  const float* bn_g    = (const float*)d_in[10];
  const float* bn_b    = (const float*)d_in[11];
  const float* bn_m    = (const float*)d_in[12];
  const float* bn_v    = (const float*)d_in[13];
  float* out = (float*)d_out;

  float* ws = (float*)d_ws;
  // region A (B*192*HW floats): qkv first, then reused for f_conv + outsum
  float* qkv    = ws;
  float* fconv  = ws;                                        // reuse after k_dw
  float* outsum = ws + (size_t)BB * 72 * HWSZ;               // still inside region A
  // region B: qkv after depthwise
  float* qkvdw  = ws + (size_t)BB * 192 * HWSZ;
  // small region C
  float* attn   = ws + (size_t)2 * BB * 192 * HWSZ;          // B*8*8*8
  float* weff   = attn + (size_t)BB * 512;                   // B*64*64

  k_qkv_gemm<<<BB * (HWSZ / 128), 256, 0, stream>>>(y, w_qkv, qkv);
  k_dw<<<(BB * 192 * HWSZ) / 256, 256, 0, stream>>>(qkv, w_dw, qkvdw);
  k_attn<<<BB * HEADS, 256, 0, stream>>>(qkvdw, temper, attn);
  k_fc_dyn<<<(BB * HWSZ) / 256, 256, 0, stream>>>(qkvdw, w_fc, b_fc, fconv);
  k_weff<<<BB, 64, 0, stream>>>(w_proj, attn, weff);
  k_groupconv<<<(BB * 64 * HWSZ) / 256, 256, 0, stream>>>(fconv, w_dep, b_dep, outsum);
  k_pv_gemm<<<BB * (HWSZ / 128), 256, 0, stream>>>(qkvdw, weff, outsum);
  k_fuse<<<BB * 64, 256, 0, stream>>>(outsum, w_fuse, y, bn_g, bn_b, bn_m, bn_v, out);
}
